// LSTM_10153302687923
// MI455X (gfx1250) — compile-verified, matrix operand order checked by reading, not run on hardware
//
#include <hip/hip_runtime.h>
#include <hip/hip_bf16.h>
#include <math.h>

// ---------- Types for WMMA ----------
typedef __attribute__((ext_vector_type(16))) _Float16 v16h;
typedef __attribute__((ext_vector_type(8)))  _Float16 h8;
typedef __attribute__((ext_vector_type(8)))  float    v8f;

#define SEQ 2048
#define BATCH 64
#define HD 256          // hidden = input = 256
#define FOURH 1024
#define NBLK 16         // persistent blocks in phase 2

// Load a 16-half WMMA A/B fragment: two contiguous 8-half (16B) segments.
// ISA 16-bit A/B layout: lane<16 -> K {k0..k0+7, k0+16..k0+23},
// lane>=16 -> K {k0+8..k0+15, k0+24..k0+31}. Caller bakes in 8*(lane>>4).
static __device__ inline v16h load_frag(const _Float16* p) {
    h8 lo = *(const h8*)p;
    h8 hi = *(const h8*)(p + 16);
    v16h r;
#pragma unroll
    for (int i = 0; i < 8; ++i) { r[i] = lo[i]; r[i + 8] = hi[i]; }
    return r;
}

static __device__ inline v8f wmma_f16(v16h a, v16h b, v8f c) {
    return __builtin_amdgcn_wmma_f32_16x16x32_f16(false, a, false, b, (short)0, c, false, false);
}

static __device__ inline float sigmoidf_(float x) { return 1.0f / (1.0f + __expf(-x)); }

// ---------- Kernel 0: f32 -> f16 convert (grid-stride) ----------
__global__ void k_cvt_f16(const float* __restrict__ s, _Float16* __restrict__ d, int n) {
    int i = blockIdx.x * blockDim.x + threadIdx.x;
    int stride = gridDim.x * blockDim.x;
    for (; i < n; i += stride) d[i] = (_Float16)s[i];
}

// ---------- Kernel 1: preact_x = X @ Wi.T + (bi+bh), f16 out ----------
// Block = 8 waves computes a 128x128 output tile. The block's Wi slice
// (128 cols x 256 k = 64KB f16, contiguous rows) is staged into LDS once;
// B fragments then come from ds_load_b128 (short latency) while the sole
// global load in the k-loop (A fragment) is double-buffered one k ahead.
__global__ void k_gemm_x(const _Float16* __restrict__ xh,
                         const _Float16* __restrict__ wih,
                         const float* __restrict__ bi,
                         const float* __restrict__ bh,
                         _Float16* __restrict__ pre) {
    extern __shared__ char smem1[];
    _Float16* ldsW = (_Float16*)smem1;           // 128 rows x 256 halfs = 64KB

    const int lane = threadIdx.x & 31;
    const int wave = threadIdx.x >> 5;
    const int mblk = blockIdx.x >> 3;            // 1024 row blocks of 128
    const int nblk = blockIdx.x & 7;             // 8 col blocks of 128
    const int koff = 8 * (lane >> 4);
    const int mrow = mblk * 128 + wave * 16 + (lane & 15);
    const _Float16* arow = xh + (size_t)mrow * HD;

    // ---- stage Wi slice: contiguous 64KB = 4096 uint4 ----
    {
        const uint4* src = ((const uint4*)wih) + (size_t)nblk * 4096;
        for (int i = threadIdx.x; i < 4096; i += 256)
            ((uint4*)ldsW)[i] = src[i];
    }
    __syncthreads();

    v8f acc[8] = {};
    v16h a_cur = load_frag(arow + koff);
#pragma unroll
    for (int kk = 0; kk < 8; ++kk) {
        const int kb = kk * 32 + koff;
        v16h a_nxt;
        if (kk < 7) a_nxt = load_frag(arow + kb + 32);   // prefetch next k-step
#pragma unroll
        for (int n = 0; n < 8; ++n) {
            v16h b = load_frag(ldsW + (size_t)(n * 16 + (lane & 15)) * HD + kb);
            acc[n] = wmma_f16(a_cur, b, acc[n]);
        }
        a_cur = a_nxt;
    }

    const int rbase = mblk * 128 + wave * 16 + 8 * (lane >> 4); // D-layout row base
#pragma unroll
    for (int n = 0; n < 8; ++n) {
        const int ncol = nblk * 128 + n * 16 + (lane & 15);
        const float bias = bi[ncol] + bh[ncol];
#pragma unroll
        for (int r = 0; r < 8; ++r)
            pre[(size_t)(rbase + r) * FOURH + ncol] = (_Float16)(acc[n][r] + bias);
    }
}

// ---------- Kernel 2: persistent recurrent phase ----------
// 16 blocks x 128 threads = 64 waves. Wave ww = blk*4+w: mtile = ww>>4
// (16 batch rows), hc = ww&15 (16 hidden cols); computes i,f,o,g tiles for
// its region -> c/h update local in registers (WMMA D-layout).
// Block preloads its Wh slice (4 gates x 64 cols x 256 k = 128KB f16) into
// LDS once; every step's B fragments come from ds_load_b128.
__global__ void k_lstm(const _Float16* __restrict__ whf,   // Wh f16 (1024,256)
                       const _Float16* __restrict__ pre,   // (SEQ*B, 1024) f16
                       const float* __restrict__ c0,       // (B,H) f32
                       _Float16* __restrict__ hbuf,        // 2 x (B*H) f16 double buffer
                       float* __restrict__ out,            // ys | h_n | c_n
                       unsigned int* __restrict__ bar) {   // SEQ counters, pre-zeroed
    extern __shared__ char smem[];
    _Float16* ldsB = (_Float16*)smem;                  // 4*64*256 halfs = 128KB
    _Float16* hA   = (_Float16*)(smem + 131072);       // 16*256 halfs = 8KB

    const int lane = threadIdx.x & 31;
    const int wave = threadIdx.x >> 5;                 // 0..3
    const int ww    = blockIdx.x * 4 + wave;           // 0..63
    const int mtile = ww >> 4;                         // 0..3 (uniform per block)
    const int hcb   = blockIdx.x & 3;                  // block's hc group (4 chunks)
    const int col   = (hcb * 4 + wave) * 16 + (lane & 15); // hidden column 0..255
    const int koff  = 8 * (lane >> 4);
    const int arow  = lane & 15;
    const int drow0 = mtile * 16 + 8 * (lane >> 4);    // D-layout batch-row base

    // ---- one-time: stage this block's Wh slice into LDS (8192 uint4) ----
    for (int i = threadIdx.x; i < 8192; i += 128) {
        const int lrow = i >> 5, u = i & 31;           // 32 uint4 per 256-half row
        const int g = lrow >> 6, lr = lrow & 63;
        ((uint4*)ldsB)[i] =
            ((const uint4*)whf)[((size_t)(g * 256 + hcb * 64 + lr) << 5) + u];
    }

    // c in D layout (8 rows per lane)
    v8f c;
#pragma unroll
    for (int r = 0; r < 8; ++r) c[r] = c0[(size_t)(drow0 + r) * HD + col];
    float hlast[8];

    __syncthreads();

    for (int t = 0; t < SEQ; ++t) {
        // prefetch this step's preact rows while we do the GEMM
        {
            const size_t p0 = (size_t)(t * BATCH + drow0) * FOURH + col;
#pragma unroll
            for (int g = 0; g < 4; ++g) __builtin_prefetch(pre + p0 + g * HD, 0, 1);
        }

        // ---- stage this block's 16 rows of h (f16, 8KB = 512 uint4) ----
        const uint4* hsrc = (const uint4*)(hbuf + (size_t)(t & 1) * (BATCH * HD) + mtile * 16 * HD);
        uint4* hdst = (uint4*)hA;
#pragma unroll
        for (int j = 0; j < 4; ++j)
            hdst[threadIdx.x + 128 * j] = hsrc[threadIdx.x + 128 * j];
        __syncthreads();

        // ---- GEMM: acc[g] += A(h tile) x B(Wh gate tile from LDS), K=256 ----
        v8f acc[4] = {};
        const _Float16* bbase = ldsB + (size_t)(wave * 16 + (lane & 15)) * HD;
#pragma unroll
        for (int kk = 0; kk < 8; ++kk) {
            const int kb = kk * 32 + koff;
            v16h a = load_frag(hA + (size_t)arow * HD + kb);
#pragma unroll
            for (int g = 0; g < 4; ++g) {
                v16h b = load_frag(bbase + (size_t)(g * 64) * HD + kb);
                acc[g] = wmma_f16(a, b, acc[g]);
            }
        }

        // ---- gates + state update (gate order: i, f, o, g) ----
#pragma unroll
        for (int r = 0; r < 8; ++r) {
            const size_t prow = (size_t)(t * BATCH + drow0 + r) * FOURH;
            float pi = acc[0][r] + (float)pre[prow + 0 * HD + col];
            float pf = acc[1][r] + (float)pre[prow + 1 * HD + col];
            float po = acc[2][r] + (float)pre[prow + 2 * HD + col];
            float pg = acc[3][r] + (float)pre[prow + 3 * HD + col];
            float it = sigmoidf_(pi), ft = sigmoidf_(pf), ot = sigmoidf_(po);
            float gt = tanhf(pg);
            c[r] = c[r] * ft + it * gt;
            float h = ot * tanhf(c[r]);
            hlast[r] = h;
            out[(size_t)(t * BATCH + drow0 + r) * HD + col] = h;               // ys
            hbuf[(size_t)((t + 1) & 1) * (BATCH * HD) + (size_t)(drow0 + r) * HD + col] = (_Float16)h;
        }

        // ---- device-wide barrier (fresh counter per step) ----
        __threadfence();
        __syncthreads();
        if (threadIdx.x == 0) {
            __hip_atomic_fetch_add(&bar[t], 1u, __ATOMIC_ACQ_REL, __HIP_MEMORY_SCOPE_AGENT);
            while (__hip_atomic_load(&bar[t], __ATOMIC_ACQUIRE, __HIP_MEMORY_SCOPE_AGENT) < (unsigned)NBLK) {}
        }
        __syncthreads();
    }

    // ---- h_n, c_n ----
    const size_t base = (size_t)SEQ * BATCH * HD;
#pragma unroll
    for (int r = 0; r < 8; ++r) {
        out[base + (size_t)(drow0 + r) * HD + col]              = hlast[r];
        out[base + BATCH * HD + (size_t)(drow0 + r) * HD + col] = c[r];
    }
}

// ---------- Host launch ----------
extern "C" void kernel_launch(void* const* d_in, const int* in_sizes, int n_in,
                              void* d_out, int out_size, void* d_ws, size_t ws_size,
                              hipStream_t stream) {
    const float* x  = (const float*)d_in[0];   // (2048,64,256)
    const float* h0 = (const float*)d_in[1];   // (1,64,256)
    const float* c0 = (const float*)d_in[2];   // (1,64,256)
    const float* Wi = (const float*)d_in[3];   // (1024,256)
    const float* bi = (const float*)d_in[4];   // (1024,)
    const float* Wh = (const float*)d_in[5];   // (1024,256)
    const float* bh = (const float*)d_in[6];   // (1024,)
    float* out = (float*)d_out;
    char* ws = (char*)d_ws;

    // Workspace layout (bytes)
    constexpr size_t OFF_BAR  = 0;                             // 2048 * 4 = 8KB
    constexpr size_t OFF_WHF  = 16384;                         // 512KB
    constexpr size_t OFF_WIF  = OFF_WHF + 524288;              // 512KB
    constexpr size_t OFF_HBUF = OFF_WIF + 524288;              // 2 * 32KB
    constexpr size_t OFF_XF   = OFF_HBUF + 65536;              // 64MB
    constexpr size_t OFF_PRE  = OFF_XF + (size_t)SEQ * BATCH * HD * 2;   // 256MB

    unsigned int* bar = (unsigned int*)(ws + OFF_BAR);
    _Float16* whf  = (_Float16*)(ws + OFF_WHF);
    _Float16* wif  = (_Float16*)(ws + OFF_WIF);
    _Float16* hbuf = (_Float16*)(ws + OFF_HBUF);
    _Float16* xf   = (_Float16*)(ws + OFF_XF);
    _Float16* pre  = (_Float16*)(ws + OFF_PRE);

    hipMemsetAsync(ws + OFF_BAR, 0, SEQ * sizeof(unsigned int), stream);

    k_cvt_f16<<<256, 256, 0, stream>>>(Wh, whf, FOURH * HD);
    k_cvt_f16<<<256, 256, 0, stream>>>(Wi, wif, FOURH * HD);
    k_cvt_f16<<<2048, 256, 0, stream>>>(x, xf, SEQ * BATCH * HD);
    k_cvt_f16<<<16, 256, 0, stream>>>(h0, hbuf, BATCH * HD);   // buffer 0 = h_{-1}

    // Phase 1: big parallel GEMM -> preact_x (f16), 128x128 tiles, 64KB LDS
    constexpr int SMEM1_BYTES = 65536;
    hipFuncSetAttribute((const void*)k_gemm_x,
                        hipFuncAttributeMaxDynamicSharedMemorySize, SMEM1_BYTES);
    k_gemm_x<<<8192, 256, SMEM1_BYTES, stream>>>(xf, wif, bi, bh, pre);

    // Phase 2: persistent recurrent kernel, 16 co-resident blocks,
    // 136KB dynamic LDS (Wh slice + h staging)
    constexpr int SMEM_BYTES = 131072 + 8192;
    hipFuncSetAttribute((const void*)k_lstm,
                        hipFuncAttributeMaxDynamicSharedMemorySize, SMEM_BYTES);
    k_lstm<<<NBLK, 128, SMEM_BYTES, stream>>>(whf, pre, c0, hbuf, out, bar);
}
